// GeneralAttention_5712306503835
// MI455X (gfx1250) — matrix operand order repaired and checked by reading, not compiled, not run on hardware
//
#include <hip/hip_runtime.h>
#include <hip/hip_bf16.h>
#include <math.h>

// ---------------------------------------------------------------------------
// Problem constants (B=2, S=2048, D=1024, H=16, DK=64)
// ---------------------------------------------------------------------------
#define BB 2
#define SS 2048
#define DD 1024
#define HH 16
#define DKK 64

typedef __bf16 bf16_t;
typedef __attribute__((ext_vector_type(16))) __bf16 v16bf;
typedef __attribute__((ext_vector_type(8)))  __bf16 v8bf;
typedef __attribute__((ext_vector_type(8)))  float  v8f;

// Load a v16bf WMMA fragment half-pair from two 16-byte-aligned addresses.
__device__ __forceinline__ v16bf ld2(const bf16_t* p0, const bf16_t* p1) {
    union { v16bf v; v8bf h[2]; } u;
    u.h[0] = *(const v8bf*)p0;
    u.h[1] = *(const v8bf*)p1;
    return u.v;
}

__device__ __forceinline__ v8f wmma_bf16(v16bf a, v16bf b, v8f c) {
    // D = A(16x32 bf16) * B(32x16 bf16) + C(16x16 f32)
    return __builtin_amdgcn_wmma_f32_16x16x32_bf16(
        /*neg_a=*/false, a, /*neg_b=*/false, b,
        /*c_mod=*/(short)0, c, /*reuse_a=*/false, /*reuse_b=*/false);
}

// ---------------------------------------------------------------------------
// CDNA5 async memory->LDS copy (ASYNCcnt-tracked), GVS addressing mode:
//   global addr = SADDR(64b, uniform) + VADDR(per-lane 32b byte offset)
//   LDS addr    = per-lane 32b byte offset in %0
// Each active lane moves 16 bytes -> one instruction moves 512 B per wave.
// ---------------------------------------------------------------------------
__device__ __forceinline__ void async_ld_b128(unsigned lds_off, unsigned gvoff,
                                              const void* sbase) {
    asm volatile("global_load_async_to_lds_b128 %0, %1, %2"
                 :
                 : "v"(lds_off), "v"(gvoff), "s"(sbase)
                 : "memory");
}
__device__ __forceinline__ void wait_async_le2() {
    asm volatile("s_wait_asynccnt 0x2" ::: "memory");
}
__device__ __forceinline__ void wait_async_0() {
    asm volatile("s_wait_asynccnt 0x0" ::: "memory");
}

// ---------------------------------------------------------------------------
// Kernel 1: RoPE on q,k (reference semantics: interleaved rotate, freq idx
// d mod 512) + bf16 conversion of q,k,v.
// ---------------------------------------------------------------------------
__global__ void rope_convert_kernel(const float* __restrict__ q,
                                    const float* __restrict__ k,
                                    const float* __restrict__ v,
                                    bf16_t* __restrict__ qr,
                                    bf16_t* __restrict__ kr,
                                    bf16_t* __restrict__ vb) {
    int idx = blockIdx.x * blockDim.x + threadIdx.x;
    if (idx >= BB * SS * (DD / 2)) return;
    int i = idx & 511;
    int s = (idx >> 9) & (SS - 1);
    int d0 = 2 * i, d1 = 2 * i + 1;

    const float LN1E4 = 9.210340371976184f;   // ln(10000)
    float invf0 = __expf(-((float)(d0 & 511) * (1.0f / 512.0f)) * LN1E4);
    float invf1 = __expf(-((float)(d1 & 511) * (1.0f / 512.0f)) * LN1E4);
    float a0 = (float)s * invf0, a1 = (float)s * invf1;
    float sn0, cs0, sn1, cs1;
    __sincosf(a0, &sn0, &cs0);
    __sincosf(a1, &sn1, &cs1);

    size_t base = ((size_t)(idx >> 9) << 10) + (size_t)d0;

    float x0 = q[base], x1 = q[base + 1];
    qr[base]     = (bf16_t)(x0 * cs0 - x1 * sn0);
    qr[base + 1] = (bf16_t)(x1 * cs1 + x0 * sn1);

    x0 = k[base]; x1 = k[base + 1];
    kr[base]     = (bf16_t)(x0 * cs0 - x1 * sn0);
    kr[base + 1] = (bf16_t)(x1 * cs1 + x0 * sn1);

    vb[base]     = (bf16_t)v[base];
    vb[base + 1] = (bf16_t)v[base + 1];
}

// ---------------------------------------------------------------------------
// Kernel 2: weight fp32 -> bf16
// ---------------------------------------------------------------------------
__global__ void wconvert_kernel(const float* __restrict__ wq,
                                const float* __restrict__ wk,
                                const float* __restrict__ wv,
                                const float* __restrict__ wo,
                                bf16_t* __restrict__ wqb, bf16_t* __restrict__ wkb,
                                bf16_t* __restrict__ wvb, bf16_t* __restrict__ wob) {
    int i = blockIdx.x * blockDim.x + threadIdx.x;
    if (i >= DD * DD) return;
    wqb[i] = (bf16_t)wq[i];
    wkb[i] = (bf16_t)wk[i];
    wvb[i] = (bf16_t)wv[i];
    wob[i] = (bf16_t)wo[i];
}

// ---------------------------------------------------------------------------
// Shared GEMM core: C(4096x1024) = X @ W^T with bf16 WMMA.
// Block = 256 thr = 8 waves = 4 m-waves x 2 col-groups; block tile 128x128;
// wave tile 32x64 (8 WMMAs / K-step). Weight tile (128 cols x 32 k) staged in
// LDS by async global->LDS b128 copies, double buffered on ASYNCcnt.
// LDS rows padded to 40 halves (80 B) -> conflict-free 16-lane b128 reads.
// NOTE: fragments are consumed one at a time (load B-frag -> 2 WMMAs) to keep
// live VGPRs under the occupancy cap — pre-materializing all fragments makes
// the allocator spill accumulators to scratch (observed in round 4).
// ---------------------------------------------------------------------------
#define WROW 40                 // padded halves per LDS weight row
#define WBUFB (128 * WROW * 2)  // bytes per LDS buffer

__device__ __forceinline__ void gemm_core_128x128(
    const bf16_t* __restrict__ A, const bf16_t* __restrict__ W,
    bf16_t (&Wlds)[2][128][WROW],
    int row0, int colb, int lane, int wave, v8f acc[2][4]) {

    const int lm = lane & 15;
    const int hi = lane >> 4;
    const int cg = wave >> 2;

    const unsigned ldsbase = (unsigned)(uintptr_t)&Wlds[0][0][0];
    // 512 16-byte chunks per (128x32) tile: chunk c -> col=c>>2, kchunk=c&3
    const int c0 = wave * 64 + lane;
    const int c1 = c0 + 32;
    const unsigned l0 = ldsbase + (unsigned)((c0 >> 2) * (WROW * 2) + (c0 & 3) * 16);
    const unsigned l1 = ldsbase + (unsigned)((c1 >> 2) * (WROW * 2) + (c1 & 3) * 16);
    const unsigned g0 = (unsigned)(((colb + (c0 >> 2)) * DD + (c0 & 3) * 8) * 2);
    const unsigned g1 = (unsigned)(((colb + (c1 >> 2)) * DD + (c1 & 3) * 8) * 2);

    // prologue: stage kb=0 into buffer 0
    async_ld_b128(l0, g0, W);
    async_ld_b128(l1, g1, W);

    for (int kb = 0; kb < DD; kb += 32) {
        const int buf = (kb >> 5) & 1;
        if (kb + 32 < DD) {
            // issue next tile into the other buffer (safe: trailing barrier of
            // the previous iteration guarantees nobody still reads it)
            const unsigned koff = (unsigned)((kb + 32) * 2);
            const unsigned lb = (unsigned)((buf ^ 1) * WBUFB);
            async_ld_b128(l0 + lb, g0 + koff, W);
            async_ld_b128(l1 + lb, g1 + koff, W);
            wait_async_le2();   // in-order: current buffer's 2 copies done
        } else {
            wait_async_0();
        }
        __syncthreads();        // all waves' copies for `buf` visible

        // A fragments for the two 16-row sub-tiles (direct global b128)
        const bf16_t* ar0 = A + (size_t)(row0 + lm) * DD + kb;
        const bf16_t* ar1 = A + (size_t)(row0 + 16 + lm) * DD + kb;
        v16bf a0 = ld2(ar0 + hi * 8, ar0 + 16 + hi * 8);
        v16bf a1 = ld2(ar1 + hi * 8, ar1 + 16 + hi * 8);
#pragma unroll
        for (int nt = 0; nt < 4; ++nt) {
            const bf16_t* bp = &Wlds[buf][cg * 64 + nt * 16 + lm][hi * 8];
            v16bf bfrag = ld2(bp, bp + 16);
            acc[0][nt] = wmma_bf16(a0, bfrag, acc[0][nt]);
            acc[1][nt] = wmma_bf16(a1, bfrag, acc[1][nt]);
        }
        __syncthreads();        // protect `buf` before it is refilled
    }
}

// ---------------------------------------------------------------------------
// Kernel 3: projection GEMMs. grid.z: 0 -> Q [b][h][s][dk], 1 -> K (same),
// 2 -> V transposed [b][h][dk][s].
// ---------------------------------------------------------------------------
__global__ void proj_gemm_kernel(const bf16_t* __restrict__ qr,
                                 const bf16_t* __restrict__ kr,
                                 const bf16_t* __restrict__ vb,
                                 const bf16_t* __restrict__ wq,
                                 const bf16_t* __restrict__ wk,
                                 const bf16_t* __restrict__ wv,
                                 bf16_t* __restrict__ Qh,
                                 bf16_t* __restrict__ Kh,
                                 bf16_t* __restrict__ Vt) {
    __shared__ __align__(16) bf16_t Wlds[2][128][WROW];

    const int z = blockIdx.z;
    const bf16_t* A = (z == 0) ? qr : ((z == 1) ? kr : vb);
    const bf16_t* W = (z == 0) ? wq : ((z == 1) ? wk : wv);

    const int lane = threadIdx.x & 31;
    const int wave = threadIdx.x >> 5;
    const int row0 = blockIdx.x * 128 + (wave & 3) * 32;
    const int colb = blockIdx.y * 128;
    const int lm = lane & 15;
    const int hi = lane >> 4;
    const int cg = wave >> 2;

    v8f acc[2][4] = {};
    gemm_core_128x128(A, W, Wlds, row0, colb, lane, wave, acc);

    // Epilogue: C layout — lanes 0-15: (M=r, N=lane); lanes 16-31: (M=r+8).
#pragma unroll
    for (int mi = 0; mi < 2; ++mi) {
#pragma unroll
        for (int nt = 0; nt < 4; ++nt) {
            int n = colb + cg * 64 + nt * 16 + lm;      // 0..1023
            int h = n >> 6, dl = n & (DKK - 1);
#pragma unroll
            for (int r = 0; r < 8; ++r) {
                int m = row0 + mi * 16 + r + hi * 8;    // 0..4095
                int b = m >> 11, s = m & (SS - 1);
                float val = acc[mi][nt][r];
                if (z < 2) {
                    bf16_t* dst = (z == 0) ? Qh : Kh;
                    dst[(((size_t)(b * HH + h)) * SS + s) * DKK + dl] = (bf16_t)val;
                } else {
                    Vt[(((size_t)(b * HH + h)) * DKK + dl) * SS + s] = (bf16_t)val;
                }
            }
        }
    }
}

// ---------------------------------------------------------------------------
// Kernel 4: causal flash attention, one wave per (b, h, 16-row q tile).
// Online softmax over 32-key steps; QK^T and P*V via bf16 WMMA.
// ---------------------------------------------------------------------------
__global__ void attn_kernel(const bf16_t* __restrict__ Qh,
                            const bf16_t* __restrict__ Kh,
                            const bf16_t* __restrict__ Vt,
                            bf16_t* __restrict__ Oc) {
    const int lane = threadIdx.x & 31;
    const int lm = lane & 15;
    const int hi = lane >> 4;
    const int bid = blockIdx.x;
    const int qt = bid & (SS / 16 - 1);
    const int h  = (bid >> 7) & (HH - 1);
    const int b  = bid >> 11;
    const int qbase = qt * 16;

    const bf16_t* Qb = Qh + ((size_t)(b * HH + h)) * SS * DKK;
    const bf16_t* Kb = Kh + ((size_t)(b * HH + h)) * SS * DKK;
    const bf16_t* Vb = Vt + ((size_t)(b * HH + h)) * DKK * SS;

    const bf16_t* qrow = Qb + (size_t)(qbase + lm) * DKK;
    v16bf qf0 = ld2(qrow + hi * 8,      qrow + 16 + hi * 8);
    v16bf qf1 = ld2(qrow + 32 + hi * 8, qrow + 48 + hi * 8);

    v8f acc[4] = {};
    float mrow[8], lrow[8];
#pragma unroll
    for (int r = 0; r < 8; ++r) { mrow[r] = -3.0e38f; lrow[r] = 0.0f; }

    __shared__ __align__(16) bf16_t Plds[16][32];   // 1 KB, wave-private

    const float scale = 0.125f;                     // 1/sqrt(DK)
    const int jend = qbase + 16;
    for (int jb = 0; jb < jend; jb += 32) {
        // ---- scores for 32 keys: preload 4 K-fragments, then 4 WMMAs ----
        const bf16_t* k0 = Kb + (size_t)(jb + lm) * DKK + hi * 8;
        const bf16_t* k1 = Kb + (size_t)(jb + 16 + lm) * DKK + hi * 8;
        v16bf kf00 = ld2(k0, k0 + 16);
        v16bf kf01 = ld2(k0 + 32, k0 + 48);
        v16bf kf10 = ld2(k1, k1 + 16);
        v16bf kf11 = ld2(k1 + 32, k1 + 48);
        v8f s0 = {}, s1 = {};
        s0 = wmma_bf16(qf0, kf00, s0);
        s1 = wmma_bf16(qf0, kf10, s1);
        s0 = wmma_bf16(qf1, kf01, s0);
        s1 = wmma_bf16(qf1, kf11, s1);

        // ---- causal mask + online softmax (row stats per C-slot) ----
        float p0[8], p1[8], corr[8];
#pragma unroll
        for (int r = 0; r < 8; ++r) {
            int qg = qbase + r + hi * 8;
            float a = s0[r] * scale;
            float c = s1[r] * scale;
            if (jb + lm > qg)      a = -1.0e9f;
            if (jb + 16 + lm > qg) c = -1.0e9f;
            float t = fmaxf(a, c);
            t = fmaxf(t, __shfl_xor(t, 1, 32));
            t = fmaxf(t, __shfl_xor(t, 2, 32));
            t = fmaxf(t, __shfl_xor(t, 4, 32));
            t = fmaxf(t, __shfl_xor(t, 8, 32));
            float mnew = fmaxf(mrow[r], t);
            corr[r] = __expf(mrow[r] - mnew);
            mrow[r] = mnew;
            a = __expf(a - mnew);
            c = __expf(c - mnew);
            p0[r] = a; p1[r] = c;
            float sum = a + c;
            sum += __shfl_xor(sum, 1, 32);
            sum += __shfl_xor(sum, 2, 32);
            sum += __shfl_xor(sum, 4, 32);
            sum += __shfl_xor(sum, 8, 32);
            lrow[r] = lrow[r] * corr[r] + sum;
        }
#pragma unroll
        for (int nt = 0; nt < 4; ++nt)
#pragma unroll
            for (int r = 0; r < 8; ++r)
                acc[nt][r] *= corr[r];

        // ---- C-layout -> A-layout relayout of P through LDS ----
#pragma unroll
        for (int r = 0; r < 8; ++r) {
            Plds[r + hi * 8][lm]      = (bf16_t)p0[r];
            Plds[r + hi * 8][16 + lm] = (bf16_t)p1[r];
        }
        __syncthreads();                 // single-wave block: cheap fence
        v16bf pf = ld2(&Plds[lm][hi * 8], &Plds[lm][16 + hi * 8]);
        __syncthreads();

        // ---- O += P(16x32) x V(32x64): per-tile load + WMMA ----
#pragma unroll
        for (int nt = 0; nt < 4; ++nt) {
            const bf16_t* vp = Vb + (size_t)(nt * 16 + lm) * SS + jb + hi * 8;
            v16bf vf = ld2(vp, vp + 16);
            acc[nt] = wmma_bf16(pf, vf, acc[nt]);
        }
    }

    // ---- normalize and store O as [b][s][h*64+d] bf16 ----
#pragma unroll
    for (int nt = 0; nt < 4; ++nt) {
#pragma unroll
        for (int r = 0; r < 8; ++r) {
            int s = qbase + r + hi * 8;
            float o = acc[nt][r] / lrow[r];
            Oc[((size_t)(b * SS + s)) * DD + h * DKK + nt * 16 + lm] = (bf16_t)o;
        }
    }
}

// ---------------------------------------------------------------------------
// Kernel 5: output projection  Y = O(4096x1024) @ Wo^T, fp32 result.
// ---------------------------------------------------------------------------
__global__ void out_gemm_kernel(const bf16_t* __restrict__ Oc,
                                const bf16_t* __restrict__ Wo,
                                float* __restrict__ Y) {
    __shared__ __align__(16) bf16_t Wlds[2][128][WROW];

    const int lane = threadIdx.x & 31;
    const int wave = threadIdx.x >> 5;
    const int row0 = blockIdx.x * 128 + (wave & 3) * 32;
    const int colb = blockIdx.y * 128;
    const int lm = lane & 15;
    const int hi = lane >> 4;
    const int cg = wave >> 2;

    v8f acc[2][4] = {};
    gemm_core_128x128(Oc, Wo, Wlds, row0, colb, lane, wave, acc);

#pragma unroll
    for (int mi = 0; mi < 2; ++mi) {
#pragma unroll
        for (int nt = 0; nt < 4; ++nt) {
            int n = colb + cg * 64 + nt * 16 + lm;
#pragma unroll
            for (int r = 0; r < 8; ++r) {
                int m = row0 + mi * 16 + r + hi * 8;
                Y[(size_t)m * DD + n] = acc[mi][nt][r];
            }
        }
    }
}

// ---------------------------------------------------------------------------
// Host launcher
// ---------------------------------------------------------------------------
extern "C" void kernel_launch(void* const* d_in, const int* in_sizes, int n_in,
                              void* d_out, int out_size, void* d_ws, size_t ws_size,
                              hipStream_t stream) {
    (void)in_sizes; (void)n_in; (void)out_size; (void)ws_size;
    const float* q  = (const float*)d_in[0];
    const float* k  = (const float*)d_in[1];
    const float* v  = (const float*)d_in[2];
    const float* wq = (const float*)d_in[3];
    const float* wk = (const float*)d_in[4];
    const float* wv = (const float*)d_in[5];
    const float* wo = (const float*)d_in[6];
    // d_in[7] = causal mask, semantics baked into attn_kernel.
    float* out = (float*)d_out;

    const size_t BSD = (size_t)BB * SS * DD;   // 4,194,304
    const size_t WSZ = (size_t)DD * DD;        // 1,048,576

    bf16_t* ws  = (bf16_t*)d_ws;
    bf16_t* qr  = ws;
    bf16_t* kr  = qr + BSD;
    bf16_t* vb  = kr + BSD;
    bf16_t* wqb = vb + BSD;
    bf16_t* wkb = wqb + WSZ;
    bf16_t* wvb = wkb + WSZ;
    bf16_t* wob = wvb + WSZ;
    bf16_t* Qh  = wob + WSZ;
    bf16_t* Kh  = Qh + BSD;
    bf16_t* Vt  = Kh + BSD;
    bf16_t* Oc  = Vt + BSD;
    // total ws use: 64 MiB of bf16

    {
        int nthr = BB * SS * (DD / 2);
        rope_convert_kernel<<<(nthr + 255) / 256, 256, 0, stream>>>(q, k, v, qr, kr, vb);
    }
    wconvert_kernel<<<(int)((WSZ + 255) / 256), 256, 0, stream>>>(wq, wk, wv, wo,
                                                                  wqb, wkb, wvb, wob);
    {
        dim3 grid((BB * SS) / 128, DD / 128, 3);
        proj_gemm_kernel<<<grid, 256, 0, stream>>>(qr, kr, vb, wqb, wkb, wvb, Qh, Kh, Vt);
    }
    attn_kernel<<<BB * HH * (SS / 16), 32, 0, stream>>>(Qh, Kh, Vt, Oc);
    {
        dim3 grid((BB * SS) / 128, DD / 128);
        out_gemm_kernel<<<grid, 256, 0, stream>>>(Oc, wob, out);
    }
}
